// SkeletonMamba_88252987998532
// MI455X (gfx1250) — matrix operand order repaired and checked by reading, not compiled, not run on hardware
//
#include <hip/hip_runtime.h>
#include <hip/hip_bf16.h>
#include <math.h>

#define B_SZ   64
#define T_SZ   8
#define J_SZ   42
#define L_SZ   337            // 1 + T*J
#define DM     192
#define DI     384
#define DSTATE 16
#define DTRANK 12
#define NXPROJ 44             // DT_RANK + 2*D_STATE
#define DTKPAD 32             // dt_proj K padded 12 -> 32
#define DEPTH  16
#define NCLS   1000
#define ML     (B_SZ * L_SZ)  // 21568

typedef __attribute__((ext_vector_type(16))) _Float16 v16h;
typedef __attribute__((ext_vector_type(8)))  _Float16 v8h;
typedef __attribute__((ext_vector_type(8)))  float    v8f;

__device__ __forceinline__ float silu_f(float x) { return x / (1.0f + __expf(-x)); }
// branchless, overflow-safe softplus: max(x,0) + log(1+exp(-|x|))
__device__ __forceinline__ float softplus_f(float x) {
  return fmaxf(x, 0.0f) + __logf(1.0f + __expf(-fabsf(x)));
}

// -------------------------------------------------------------------------
// Embedding. 2D grid: blockIdx.x = l, blockIdx.y = b, threadIdx.x = d.
// -------------------------------------------------------------------------
__global__ void embed_kernel(const float* __restrict__ x,
                             const float* __restrict__ jw,
                             const float* __restrict__ jb,
                             const float* __restrict__ cls,
                             const float* __restrict__ spos,
                             const float* __restrict__ tpos,
                             float* __restrict__ h)
{
  int d = threadIdx.x;
  int l = blockIdx.x;
  int b = blockIdx.y;
  float v;
  if (l == 0) {
    v = cls[d] + spos[d];
  } else {
    int t = (l - 1) / J_SZ, j = (l - 1) % J_SZ;   // block-uniform -> SALU
    const float* xp = x + (((size_t)b * T_SZ + t) * J_SZ + j) * 3;
    v = xp[0] * jw[d * 3 + 0] + xp[1] * jw[d * 3 + 1] + xp[2] * jw[d * 3 + 2]
        + jb[d] + spos[(j + 1) * DM + d] + tpos[t * DM + d];
  }
  h[((size_t)b * L_SZ + l) * DM + d] = v;
}

// -------------------------------------------------------------------------
// Weight convert + pad (f32 [nl,R,K] -> f16 [nl,Rpad,Kpad], zero-filled)
// -------------------------------------------------------------------------
__global__ void cvt_pad_kernel(const float* __restrict__ src,
                               _Float16* __restrict__ dst,
                               int nl, int R, int Rpad, int K, int Kpad)
{
  int idx = blockIdx.x * blockDim.x + threadIdx.x;
  int tot = nl * Rpad * Kpad;
  if (idx >= tot) return;
  int k = idx % Kpad;
  int r = (idx / Kpad) % Rpad;
  int l = idx / (Kpad * Rpad);
  float v = (r < R && k < K) ? src[((size_t)l * R + r) * K + k] : 0.0f;
  dst[idx] = (_Float16)v;
}

// dt slice convert: dt16[row, 0:32] from dbl[row, 0:12], zero-padded
__global__ void dt_cvt_kernel(const float* __restrict__ dbl,
                              _Float16* __restrict__ dt16)
{
  int idx = blockIdx.x * blockDim.x + threadIdx.x;
  if (idx >= ML * DTKPAD) return;
  int k = idx & (DTKPAD - 1);
  int row = idx >> 5;
  float v = (k < DTRANK) ? dbl[(size_t)row * NXPROJ + k] : 0.0f;
  dt16[idx] = (_Float16)v;
}

// -------------------------------------------------------------------------
// residual += h (or = h if first), hn16 = f16(rmsnorm(residual) * w)
// -------------------------------------------------------------------------
__global__ void resnorm_kernel(const float* __restrict__ hin,
                               float* __restrict__ residual,
                               _Float16* __restrict__ hn16,
                               const float* __restrict__ w, int first)
{
  __shared__ float sred[64];
  __shared__ float sscale;
  int row = blockIdx.x;
  int tid = threadIdx.x;
  size_t base = (size_t)row * DM;
  float r0 = hin[base + tid], r1 = hin[base + tid + 64], r2 = hin[base + tid + 128];
  if (!first) {
    r0 += residual[base + tid];
    r1 += residual[base + tid + 64];
    r2 += residual[base + tid + 128];
  }
  residual[base + tid]       = r0;
  residual[base + tid + 64]  = r1;
  residual[base + tid + 128] = r2;
  sred[tid] = r0 * r0 + r1 * r1 + r2 * r2;
  __syncthreads();
  if (tid < 32) {
    float v = sred[tid] + sred[tid + 32];
    for (int o = 16; o > 0; o >>= 1) v += __shfl_down(v, o, 32);
    if (tid == 0) sscale = rsqrtf(v / (float)DM + 1e-5f);
  }
  __syncthreads();
  float sc = sscale;
  hn16[base + tid]       = (_Float16)(r0 * sc * w[tid]);
  hn16[base + tid + 64]  = (_Float16)(r1 * sc * w[tid + 64]);
  hn16[base + tid + 128] = (_Float16)(r2 * sc * w[tid + 128]);
}

// -------------------------------------------------------------------------
// Final norm: feat16[b] = f16(rmsnorm(h[b,0,:] + residual[b,0,:]) * norm_f_w)
// -------------------------------------------------------------------------
__global__ void final_norm_kernel(const float* __restrict__ hin,
                                  const float* __restrict__ residual,
                                  const float* __restrict__ w,
                                  _Float16* __restrict__ feat16)
{
  __shared__ float sred[64];
  __shared__ float sscale;
  int b = blockIdx.x;
  int tid = threadIdx.x;
  size_t base = (size_t)b * L_SZ * DM;   // token 0
  float r0 = hin[base + tid]       + residual[base + tid];
  float r1 = hin[base + tid + 64]  + residual[base + tid + 64];
  float r2 = hin[base + tid + 128] + residual[base + tid + 128];
  sred[tid] = r0 * r0 + r1 * r1 + r2 * r2;
  __syncthreads();
  if (tid < 32) {
    float v = sred[tid] + sred[tid + 32];
    for (int o = 16; o > 0; o >>= 1) v += __shfl_down(v, o, 32);
    if (tid == 0) sscale = rsqrtf(v / (float)DM + 1e-5f);
  }
  __syncthreads();
  float sc = sscale;
  feat16[(size_t)b * DM + tid]       = (_Float16)(r0 * sc * w[tid]);
  feat16[(size_t)b * DM + tid + 64]  = (_Float16)(r1 * sc * w[tid + 64]);
  feat16[(size_t)b * DM + tid + 128] = (_Float16)(r2 * sc * w[tid + 128]);
}

// -------------------------------------------------------------------------
// WMMA GEMM, f16 operands, f32 accumulate, K fully unrolled (template KPAD).
//   C[M,N] = act( A16[M,KPAD] * W16[Npad,KPAD]^T + bias )
// The 16xKPAD A tile (shared by all 4 waves of the block) is staged in LDS
// via CDNA5 async-to-LDS DMA (global_load_async_to_lds_b128 / ASYNCcnt);
// LDS rows padded by 8 halves so lane reads spread across all 64 banks.
// B is streamed from global as contiguous b128 clauses.
// act: 0 = none, 1 = softplus
// -------------------------------------------------------------------------
template <int KPAD>
__global__ void gemm_wmma_kernel(const _Float16* __restrict__ A,
                                 const _Float16* __restrict__ W,
                                 const float* __restrict__ bias,
                                 float* __restrict__ C, int ldc,
                                 int N, int act)
{
  constexpr int LROW = KPAD + 8;           // LDS row stride (halves)
  __shared__ _Float16 sA[16 * LROW];

  const int tid   = threadIdx.x;
  const int lane  = tid & 31;
  const int wave  = tid >> 5;
  const int tileN = (blockIdx.x * 4 + wave) * 16;
  const int tileM = blockIdx.y * 16;
  const int Npad16 = (N + 15) & ~15;

  // ---- cooperative async DMA of A tile into LDS (16 rows x KPAD halves) ----
  {
    constexpr int NT = 16 * KPAD / 8;      // number of 16-byte transfers
    constexpr int CPR = KPAD / 8;          // 16-byte chunks per row
    const char* gsrc = (const char*)(A + (size_t)tileM * KPAD);
    unsigned ldsBase = (unsigned)(size_t)sA;
#pragma unroll
    for (int r = 0; r < (NT + 127) / 128; ++r) {
      int idx = tid + r * 128;
      if (idx < NT) {
        int row = idx / CPR;
        int c16 = idx % CPR;
        unsigned lds = ldsBase + (unsigned)(row * LROW * 2 + c16 * 16);
        unsigned long long g = (unsigned long long)(gsrc + (size_t)row * (KPAD * 2) + c16 * 16);
        asm volatile("global_load_async_to_lds_b128 %0, %1, off"
                     :: "v"(lds), "v"(g) : "memory");
      }
    }
    asm volatile("s_wait_asynccnt 0" ::: "memory");
  }
  __syncthreads();

  if (tileN >= Npad16) return;   // wave-uniform; EXEC all-1 for WMMA

  const int rowL  = lane & 15;
  const int colB  = tileN + (lane & 15);
  const int khalf = lane >> 4;

  const _Float16* As = sA + rowL * LROW + khalf * 8;
  const _Float16* Wp = W + (size_t)colB * KPAD + khalf * 8;

  v8f acc = {};
#pragma unroll
  for (int kk = 0; kk < KPAD; kk += 32) {
    v8h  alo = *(const v8h*)(As + kk);          // ds_load_b128
    v8h  ahi = *(const v8h*)(As + kk + 16);
    v8h  blo = *(const v8h*)(Wp + kk);          // 2x global_load_b128
    v8h  bhi = *(const v8h*)(Wp + kk + 16);
    v16h b   = __builtin_shufflevector(blo, bhi,
                                       0, 1, 2, 3, 4, 5, 6, 7,
                                       8, 9, 10, 11, 12, 13, 14, 15);
    v16h a   = __builtin_shufflevector(alo, ahi,
                                       0, 1, 2, 3, 4, 5, 6, 7,
                                       8, 9, 10, 11, 12, 13, 14, 15);
    acc = __builtin_amdgcn_wmma_f32_16x16x32_f16(false, a, false, b,
                                                 (short)0, acc, false, false);
  }

  // ---- epilogue: bias hoisted (c is loop-invariant); M always mult of 16 ----
  const int c = tileN + (lane & 15);
  const bool cok = (c < N);
  float bv = 0.0f;
  if (bias && cok) bv = bias[c];
#pragma unroll
  for (int v = 0; v < 8; ++v) {
    int r = tileM + v + (khalf << 3);
    float val = acc[v] + bv;
    if (act == 1) val = softplus_f(val);   // branchless, no divergence
    if (cok) C[(size_t)r * ldc + c] = val;
  }
}

// -------------------------------------------------------------------------
// Depthwise causal conv (width 4) + bias + SiLU; dual-writes f32 + f16.
// 2D grid: blockIdx.x = l, blockIdx.y = b, threadIdx.x = d.
// -------------------------------------------------------------------------
__global__ void conv_silu_kernel(const float* __restrict__ xz,
                                 const float* __restrict__ cw,
                                 const float* __restrict__ cb,
                                 float* __restrict__ u,
                                 _Float16* __restrict__ u16)
{
  int d = threadIdx.x;
  int l = blockIdx.x;
  int b = blockIdx.y;
  const float* w = cw + d * 4;
  float acc = cb[d];
#pragma unroll
  for (int c = 0; c < 4; ++c) {
    int ls = l - 3 + c;
    if (ls >= 0) acc += w[c] * xz[((size_t)b * L_SZ + ls) * (2 * DI) + d];
  }
  float s = silu_f(acc);
  size_t idx = ((size_t)b * L_SZ + l) * DI + d;
  u[idx]   = s;
  u16[idx] = (_Float16)s;
}

// -------------------------------------------------------------------------
// Selective scan. One block per batch b (384 threads = d channels).
// 16 recurrent states/thread; B/C vectors staged in LDS per step.
// Fuses  y = (scan + u*Dp) * silu(z); writes f16 for out_proj GEMM.
// -------------------------------------------------------------------------
__global__ void scan_kernel(const float* __restrict__ delta,
                            const float* __restrict__ xz,      // for z
                            const float* __restrict__ u,
                            const float* __restrict__ dbl,     // [ML,44]; Bm@12, Cm@28
                            const float* __restrict__ A_log,   // [DI,16] layer slice
                            const float* __restrict__ Dp,      // [DI]
                            _Float16* __restrict__ y16)
{
  __shared__ float sB[DSTATE];
  __shared__ float sC[DSTATE];
  int b = blockIdx.x;
  int d = threadIdx.x;

  float a[DSTATE];
#pragma unroll
  for (int s = 0; s < DSTATE; ++s) a[s] = -__expf(A_log[d * DSTATE + s]);
  float Dv = Dp[d];

  float h[DSTATE];
#pragma unroll
  for (int s = 0; s < DSTATE; ++s) h[s] = 0.0f;

  for (int l = 0; l < L_SZ; ++l) {
    size_t row = (size_t)b * L_SZ + l;
    if (d < DSTATE)          sB[d]          = dbl[row * NXPROJ + DTRANK + d];
    else if (d < 2 * DSTATE) sC[d - DSTATE] = dbl[row * NXPROJ + DTRANK + d];
    __syncthreads();

    float dt = delta[row * DI + d];
    float uv = u[row * DI + d];
    if (l + 1 < L_SZ) {   // gfx1250 global_prefetch for next step
      __builtin_prefetch(delta + (row + 1) * DI + d, 0, 3);
      __builtin_prefetch(u + (row + 1) * DI + d, 0, 3);
    }
    float du = dt * uv;
    float yv = 0.0f;
#pragma unroll
    for (int s = 0; s < DSTATE; ++s) {
      h[s] = h[s] * __expf(dt * a[s]) + du * sB[s];
      yv += h[s] * sC[s];
    }
    float zv = xz[row * (2 * DI) + DI + d];
    y16[row * DI + d] = (_Float16)((yv + uv * Dv) * silu_f(zv));
    __syncthreads();
  }
}

// -------------------------------------------------------------------------
// Host launcher
// -------------------------------------------------------------------------
template <int KPAD>
static inline void launch_gemm(const _Float16* A, const _Float16* W,
                               const float* bias, float* C, int ldc,
                               int M, int N, int act, hipStream_t stream)
{
  int Npad16 = (N + 15) & ~15;
  dim3 grid((Npad16 + 63) / 64, (M + 15) / 16);
  gemm_wmma_kernel<KPAD><<<grid, 128, 0, stream>>>(A, W, bias, C, ldc, N, act);
}

extern "C" void kernel_launch(void* const* d_in, const int* in_sizes, int n_in,
                              void* d_out, int out_size, void* d_ws, size_t ws_size,
                              hipStream_t stream)
{
  (void)in_sizes; (void)n_in; (void)out_size; (void)ws_size;

  const float* x         = (const float*)d_in[0];
  const float* jw        = (const float*)d_in[1];
  const float* jb        = (const float*)d_in[2];
  const float* cls       = (const float*)d_in[3];
  const float* spos      = (const float*)d_in[4];
  const float* tpos      = (const float*)d_in[5];
  const float* norm_w    = (const float*)d_in[6];
  const float* in_proj_w = (const float*)d_in[7];
  const float* conv_w    = (const float*)d_in[8];
  const float* conv_b    = (const float*)d_in[9];
  const float* x_proj_w  = (const float*)d_in[10];
  const float* dt_proj_w = (const float*)d_in[11];
  const float* dt_proj_b = (const float*)d_in[12];
  const float* A_log     = (const float*)d_in[13];
  const float* Dp        = (const float*)d_in[14];
  const float* out_proj_w= (const float*)d_in[15];
  const float* norm_f_w  = (const float*)d_in[16];
  const float* head_w    = (const float*)d_in[17];
  const float* head_b    = (const float*)d_in[18];
  float* out = (float*)d_out;

  // ---- workspace carve-out: f32 region first, then f16 region ----
  float* wsf      = (float*)d_ws;
  float* residual = wsf;  wsf += (size_t)ML * DM;
  float* hbuf     = wsf;  wsf += (size_t)ML * DM;
  float* xz       = wsf;  wsf += (size_t)ML * 2 * DI;
  float* u        = wsf;  wsf += (size_t)ML * DI;
  float* dbl      = wsf;  wsf += (size_t)ML * NXPROJ;
  float* delta    = wsf;  wsf += (size_t)ML * DI;

  _Float16* wsh   = (_Float16*)wsf;
  _Float16* hn16  = wsh;  wsh += (size_t)ML * DM;
  _Float16* u16   = wsh;  wsh += (size_t)ML * DI;
  _Float16* dt16  = wsh;  wsh += (size_t)ML * DTKPAD;
  _Float16* y16   = wsh;  wsh += (size_t)ML * DI;
  _Float16* feat16= wsh;  wsh += (size_t)B_SZ * DM;
  _Float16* w_in16  = wsh;  wsh += (size_t)DEPTH * 2 * DI * DM;   // [16,768,192]
  _Float16* w_xp16  = wsh;  wsh += (size_t)DEPTH * 48 * DI;       // [16,48,384] (44->48)
  _Float16* w_dt16  = wsh;  wsh += (size_t)DEPTH * DI * DTKPAD;   // [16,384,32] (12->32)
  _Float16* w_out16 = wsh;  wsh += (size_t)DEPTH * DM * DI;       // [16,192,384]
  _Float16* w_hd16  = wsh;  wsh += (size_t)1008 * DM;             // [1008,192] (1000->1008)

  // ---- weight conversion (cheap; per launch for determinism) ----
  {
    int t;
    t = DEPTH * 2 * DI * DM;
    cvt_pad_kernel<<<(t + 255) / 256, 256, 0, stream>>>(in_proj_w, w_in16, DEPTH, 2 * DI, 2 * DI, DM, DM);
    t = DEPTH * 48 * DI;
    cvt_pad_kernel<<<(t + 255) / 256, 256, 0, stream>>>(x_proj_w, w_xp16, DEPTH, NXPROJ, 48, DI, DI);
    t = DEPTH * DI * DTKPAD;
    cvt_pad_kernel<<<(t + 255) / 256, 256, 0, stream>>>(dt_proj_w, w_dt16, DEPTH, DI, DI, DTRANK, DTKPAD);
    t = DEPTH * DM * DI;
    cvt_pad_kernel<<<(t + 255) / 256, 256, 0, stream>>>(out_proj_w, w_out16, DEPTH, DM, DM, DI, DI);
    t = 1008 * DM;
    cvt_pad_kernel<<<(t + 255) / 256, 256, 0, stream>>>(head_w, w_hd16, 1, NCLS, 1008, DM, DM);
  }

  // ---- embedding ----
  {
    dim3 g(L_SZ, B_SZ);
    embed_kernel<<<g, DM, 0, stream>>>(x, jw, jb, cls, spos, tpos, hbuf);
  }

  // ---- 16 Mamba layers ----
  for (int layer = 0; layer < DEPTH; ++layer) {
    resnorm_kernel<<<ML, 64, 0, stream>>>(hbuf, residual, hn16,
                                          norm_w + (size_t)layer * DM, layer == 0);

    // in_proj: [ML,192] x [768,192]^T -> xz [ML,768]
    launch_gemm<DM>(hn16, w_in16 + (size_t)layer * 2 * DI * DM, nullptr,
                    xz, 2 * DI, ML, 2 * DI, 0, stream);

    // depthwise causal conv + silu -> u (f32) and u16
    {
      dim3 g(L_SZ, B_SZ);
      conv_silu_kernel<<<g, DI, 0, stream>>>(
          xz, conv_w + (size_t)layer * DI * 4, conv_b + (size_t)layer * DI, u, u16);
    }

    // x_proj: [ML,384] x [44,384]^T -> dbl [ML,44]
    launch_gemm<DI>(u16, w_xp16 + (size_t)layer * 48 * DI, nullptr,
                    dbl, NXPROJ, ML, NXPROJ, 0, stream);

    // dt slice -> dt16 [ML,32] (zero-padded)
    {
      int tot = ML * DTKPAD;
      dt_cvt_kernel<<<(tot + 255) / 256, 256, 0, stream>>>(dbl, dt16);
    }

    // dt_proj + bias + softplus -> delta [ML,384]
    launch_gemm<DTKPAD>(dt16, w_dt16 + (size_t)layer * DI * DTKPAD,
                        dt_proj_b + (size_t)layer * DI,
                        delta, DI, ML, DI, 1, stream);

    // selective scan (fused +u*Dp, *silu(z)) -> y16 [ML,384]
    scan_kernel<<<B_SZ, DI, 0, stream>>>(delta, xz, u, dbl,
                                         A_log + (size_t)layer * DI * DSTATE,
                                         Dp + (size_t)layer * DI, y16);

    // out_proj: [ML,384] x [192,384]^T -> hbuf [ML,192]
    launch_gemm<DI>(y16, w_out16 + (size_t)layer * DM * DI, nullptr,
                    hbuf, DM, ML, DM, 0, stream);
  }

  // ---- final norm on token 0 + head ----
  final_norm_kernel<<<B_SZ, 64, 0, stream>>>(hbuf, residual, norm_f_w, feat16);
  launch_gemm<DM>(feat16, w_hd16, head_b, out, NCLS, B_SZ, NCLS, 0, stream);
}
